// SpaceTimeBlock_10453950398737
// MI455X (gfx1250) — matrix operand order repaired and checked by reading, not compiled, not run on hardware
//
#include <hip/hip_runtime.h>

// ---------------------------------------------------------------------------
// Problem constants
// ---------------------------------------------------------------------------
static constexpr long D      = 768;
static constexpr long NT     = 3136;          // 392 * 8   time-mamba tokens
static constexpr long NS     = 3152;          // 16 * 197  space-attn tokens
static constexpr long NX     = 3138;          // 2 * 1569  full tokens
static constexpr long NHEAD  = 12;
static constexpr long HD     = 64;
static constexpr long LSEQ   = 197;
static constexpr long NPAIR  = 16;

// ---------------------------------------------------------------------------
// Workspace layout (byte offsets, 256B aligned)
// ---------------------------------------------------------------------------
static constexpr size_t alup(size_t x) { return (x + 255) & ~(size_t)255; }
// fp32 buffers
static constexpr size_t B_XT   = 0;
static constexpr size_t B_XZ   = alup(B_XT   + NT * D * 4);         // [NT][1536] f32
static constexpr size_t B_XCF  = alup(B_XZ   + NT * 1536 * 4);
static constexpr size_t B_XCB  = alup(B_XCF  + NT * D * 4);
static constexpr size_t B_XDF  = alup(B_XCB  + NT * D * 4);         // [NT][80] f32
static constexpr size_t B_XDB  = alup(B_XDF  + NT * 80 * 4);
static constexpr size_t B_DLF  = alup(B_XDB  + NT * 80 * 4);
static constexpr size_t B_DLB  = alup(B_DLF  + NT * D * 4);
static constexpr size_t B_YF   = alup(B_DLB  + NT * D * 4);
static constexpr size_t B_YB   = alup(B_YF   + NT * D * 4);
static constexpr size_t B_TOUT = alup(B_YB   + NT * D * 4);
static constexpr size_t B_TRES = alup(B_TOUT + NT * D * 4);
static constexpr size_t B_XS   = alup(B_TRES + NT * D * 4);         // [16][197][768]
static constexpr size_t B_SC   = alup(B_XS   + NS * D * 4);         // [192][197][197]
static constexpr size_t B_SOUT = alup(B_SC   + 192L * LSEQ * LSEQ * 4);
static constexpr size_t B_XNEW = alup(B_SOUT + NS * D * 4);
static constexpr size_t B_H2   = alup(B_XNEW + NX * D * 4);
// bf16 buffers (ushort)
static constexpr size_t B_LN3B = alup(B_H2   + NX * D * 4);
static constexpr size_t B_XCFB = alup(B_LN3B + NT * D * 2);
static constexpr size_t B_XCBB = alup(B_XCFB + NT * D * 2);
static constexpr size_t B_XDFB = alup(B_XCBB + NT * D * 2);
static constexpr size_t B_XDBB = alup(B_XDFB + NT * 80 * 2);
static constexpr size_t B_YCB  = alup(B_XDBB + NT * 80 * 2);
static constexpr size_t B_LN1B = alup(B_YCB  + NT * D * 2);
static constexpr size_t B_QKVB = alup(B_LN1B + NS * D * 2);         // [16][197][2304]
static constexpr size_t B_PB   = alup(B_QKVB + NS * 2304 * 2);      // [192][197][197]
static constexpr size_t B_AOB  = alup(B_PB   + 192L * LSEQ * LSEQ * 2);
static constexpr size_t B_LN2B = alup(B_AOB  + NS * D * 2);
static constexpr size_t B_H1B  = alup(B_LN2B + NX * D * 2);         // [NX][3072]
// bf16 weight copies
static constexpr size_t B_WIN  = alup(B_H1B  + NX * 3072 * 2);
static constexpr size_t B_WXF  = alup(B_WIN  + 1536L * 768 * 2);
static constexpr size_t B_WXB  = alup(B_WXF  + 80L * 768 * 2);
static constexpr size_t B_WDF  = alup(B_WXB  + 80L * 768 * 2);
static constexpr size_t B_WDB  = alup(B_WDF  + 768L * 48 * 2);
static constexpr size_t B_WOW  = alup(B_WDB  + 768L * 48 * 2);
static constexpr size_t B_WQKV = alup(B_WOW  + 768L * 768 * 2);
static constexpr size_t B_WAO  = alup(B_WQKV + 2304L * 768 * 2);
static constexpr size_t B_WFC1 = alup(B_WAO  + 768L * 768 * 2);
static constexpr size_t B_WFC2 = alup(B_WFC1 + 3072L * 768 * 2);

// ---------------------------------------------------------------------------
// WMMA types / helpers
// ---------------------------------------------------------------------------
typedef __bf16          bf16_t;
typedef bf16_t          v16bf __attribute__((ext_vector_type(16)));
typedef float           v8f   __attribute__((ext_vector_type(8)));
typedef unsigned short  us4   __attribute__((ext_vector_type(4)));
typedef unsigned short  us8   __attribute__((ext_vector_type(8)));
typedef unsigned short  us16  __attribute__((ext_vector_type(16)));

__device__ __forceinline__ unsigned short f2bf(float f) {
  unsigned u = __builtin_bit_cast(unsigned, f);
  unsigned r = u + 0x7FFFu + ((u >> 16) & 1u);   // round-to-nearest-even
  return (unsigned short)(r >> 16);
}
__device__ __forceinline__ float siluf(float x)  { return x / (1.0f + __expf(-x)); }
__device__ __forceinline__ float geluf(float x)  { return 0.5f * x * (1.0f + erff(x * 0.70710678118654752f)); }
__device__ __forceinline__ float softplusf(float x) { return (x > 20.0f) ? x : log1pf(__expf(x)); }

// async 16B copy: global -> LDS, no VGPR round trip (ASYNCcnt-tracked)
__device__ __forceinline__ void async_cp16(unsigned short* lds_p,
                                           const unsigned short* gp) {
  unsigned lofs = (unsigned)(unsigned long long)lds_p;   // low 32b = LDS offset
  asm volatile("global_load_async_to_lds_b128 %0, %1, off"
               :: "v"(lofs), "v"(gp) : "memory");
}
__device__ __forceinline__ void wait_async0() {
  asm volatile("s_wait_asynccnt 0x0" ::: "memory");
}

// ---------------------------------------------------------------------------
// bf16-in / bf16-or-f32-out WMMA GEMM:
//   C = act( scale * A[M][K](bf16) * op(B)(bf16) + bias )
//   BT=true : B stored [N][K];  BT=false: B stored [K][N]
//   OUT: 0 = f32 C, 1 = bf16 C16, 2 = both
// Block tile 128x128, 8 waves of 32x64 (8 WMMA/wave/K-step),
// LDS double-buffered; fast-path staging uses GLOBAL_LOAD_ASYNC_TO_LDS_B128.
// ACT: 0=none, 1=+bias, 2=+bias,gelu, 3=+bias,softplus
// ---------------------------------------------------------------------------
#define TMM 128
#define TK  32
#define TST 40   // ushort row stride (80B, 16B aligned)

template <int ACT, bool BT, int OUT>
__global__ __launch_bounds__(256) void k_gemm(
    const unsigned short* __restrict__ A, int lda, long ApS, long AhS,
    const unsigned short* __restrict__ B, int ldb, long BpS, long BhS,
    float* __restrict__ C, unsigned short* __restrict__ C16,
    int ldc, long CpS, long ChS,
    const float* __restrict__ bias,
    int M, int N, int K, int H, float scale)
{
  __shared__ unsigned short At[2][TMM * TST];
  __shared__ unsigned short Bt[2][TMM * TST];

  const int tid  = threadIdx.x;
  const int lane = tid & 31;
  const int wave = tid >> 5;
  const int wm   = (wave >> 1) * 32;   // 4 waves along M
  const int wn   = (wave & 1) * 64;    // 2 waves along N

  const int pair = blockIdx.z / H, hh = blockIdx.z % H;
  const unsigned short* Ab = A + (long)pair * ApS + (long)hh * AhS;
  const unsigned short* Bb = B + (long)pair * BpS + (long)hh * BhS;
  const long cOff = (long)pair * CpS + (long)hh * ChS;

  const int m0 = blockIdx.y * TMM;
  const int n0 = blockIdx.x * TMM;

  const bool fullM  = (m0 + TMM <= M);
  const bool fullN  = (n0 + TMM <= N);
  const bool ldaVec = ((lda & 7) == 0);
  const bool ldbVec = ((ldb & 7) == 0);

  // stage A tile (128 x 32 bf16) into buffer `bi` (16 ushort/thread)
  auto stageA = [&](int bi, int kt) {
    if (fullM && (kt + TK <= K) && ldaVec) {
#pragma unroll
      for (int i = 0; i < 2; ++i) {
        int e = (i * 256 + tid) * 8;
        int r = e >> 5, c = e & 31;
        async_cp16(&At[bi][r * TST + c], Ab + (long)(m0 + r) * lda + (kt + c));
      }
    } else {
#pragma unroll
      for (int i = 0; i < 4; ++i) {
        int e = (i * 256 + tid) * 4;
        int r = e >> 5, c = e & 31;
        int gr = m0 + r;
        us4 vv;
#pragma unroll
        for (int q = 0; q < 4; ++q) {
          int gc = kt + c + q;
          bool ok = (gr < M) && (gc < K);
          unsigned short t = Ab[ok ? ((long)gr * lda + gc) : 0];  // clamped
          vv[q] = ok ? t : (unsigned short)0;
        }
        *(us4*)&At[bi][r * TST + c] = vv;
      }
    }
  };
  // stage B tile (128 x 32 bf16) into buffer `bi`, Bt[n][k]
  auto stageB = [&](int bi, int kt) {
    if (BT && fullN && (kt + TK <= K) && ldbVec) {
#pragma unroll
      for (int i = 0; i < 2; ++i) {
        int e = (i * 256 + tid) * 8;
        int r = e >> 5, c = e & 31;
        async_cp16(&Bt[bi][r * TST + c], Bb + (long)(n0 + r) * ldb + (kt + c));
      }
    } else {
#pragma unroll
      for (int i = 0; i < 4; ++i) {
        int e = (i * 256 + tid) * 4;
        int r = e >> 5, c = e & 31;
        int gn = n0 + r;
        us4 vv;
#pragma unroll
        for (int q = 0; q < 4; ++q) {
          int gk = kt + c + q;
          bool ok = (gn < N) && (gk < K);
          long off = BT ? ((long)gn * ldb + gk) : ((long)gk * ldb + gn);
          unsigned short t = Bb[ok ? off : 0];
          vv[q] = ok ? t : (unsigned short)0;
        }
        *(us4*)&Bt[bi][r * TST + c] = vv;
      }
    }
  };

  v8f acc[2][4];
#pragma unroll
  for (int i = 0; i < 2; ++i)
#pragma unroll
    for (int j = 0; j < 4; ++j)
#pragma unroll
      for (int r = 0; r < 8; ++r) acc[i][j][r] = 0.0f;

  const int rsel = lane & 15;
  const int kbA  = (lane < 16) ? 0 : 8;    // A frag: K {kb..kb+7, kb+16..kb+23}
  const int kbB  = (lane < 16) ? 0 : 8;    // B frag: K {kb..kb+7, kb+16..kb+23}

  stageA(0, 0);
  stageB(0, 0);
  wait_async0();
  __syncthreads();

  int buf = 0;
  for (int kt = 0; kt < K; kt += TK) {
    if (kt + TK < K) { stageA(buf ^ 1, kt + TK); stageB(buf ^ 1, kt + TK); }

    v16bf afrag[2], bfrag[4];
#pragma unroll
    for (int t = 0; t < 2; ++t) {
      int row = wm + t * 16 + rsel;
      us8 lo = *(const us8*)&At[buf][row * TST + kbA];
      us8 hi = *(const us8*)&At[buf][row * TST + kbA + 16];
      us16 u = __builtin_shufflevector(lo, hi, 0,1,2,3,4,5,6,7,8,9,10,11,12,13,14,15);
      afrag[t] = __builtin_bit_cast(v16bf, u);
    }
#pragma unroll
    for (int t = 0; t < 4; ++t) {
      int col = wn + t * 16 + rsel;
      us8 blo = *(const us8*)&Bt[buf][col * TST + kbB];
      us8 bhi = *(const us8*)&Bt[buf][col * TST + kbB + 16];
      us16 bu = __builtin_shufflevector(blo, bhi, 0,1,2,3,4,5,6,7,8,9,10,11,12,13,14,15);
      bfrag[t] = __builtin_bit_cast(v16bf, bu);
    }
#pragma unroll
    for (int i = 0; i < 2; ++i)
#pragma unroll
      for (int j = 0; j < 4; ++j)
        acc[i][j] = __builtin_amdgcn_wmma_f32_16x16x32_bf16(
            false, afrag[i], false, bfrag[j], (short)0, acc[i][j], false, false);

    wait_async0();
    __syncthreads();
    buf ^= 1;
  }

  // ---- epilogue (f32 C/D layout: VGPR r -> M = r (+8 for upper half-wave)) ----
#pragma unroll
  for (int i = 0; i < 2; ++i)
#pragma unroll
    for (int j = 0; j < 4; ++j) {
      int gn = n0 + wn + j * 16 + (lane & 15);
      int mb = m0 + wm + i * 16 + ((lane < 16) ? 0 : 8);
      bool cf = fullM && fullN;
#pragma unroll
      for (int r = 0; r < 8; ++r) {
        int gm = mb + r;
        if (cf || (gm < M && gn < N)) {
          float v = acc[i][j][r] * scale;
          if (ACT >= 1) v += bias[gn];
          if (ACT == 2) v = geluf(v);
          if (ACT == 3) v = softplusf(v);
          long o = cOff + (long)gm * ldc + gn;
          if (OUT == 0 || OUT == 2) C[o] = v;
          if (OUT == 1 || OUT == 2) C16[o] = f2bf(v);
        }
      }
    }
}

// ---------------------------------------------------------------------------
// Elementwise / reduction kernels
// ---------------------------------------------------------------------------
__global__ __launch_bounds__(256) void k_cvt(const float* __restrict__ in,
                                             unsigned short* __restrict__ o, long n) {
  long idx = (long)blockIdx.x * 256 + threadIdx.x;
  if (idx < n) o[idx] = f2bf(in[idx]);
}

__global__ __launch_bounds__(256) void k_gather_xt(const float* __restrict__ x,
                                                   float* __restrict__ xt) {
  long idx = (long)blockIdx.x * 256 + threadIdx.x;
  if (idx >= NT * D) return;
  long tok = idx / D; int d = idx % D;
  long b = tok / 1568, s = tok % 1568;
  xt[idx] = x[(b * 1569 + 1 + s) * D + d];
}

// LayerNorm -> bf16 output (all LN outputs feed GEMMs)
__global__ __launch_bounds__(256) void k_layernorm(const float* __restrict__ x,
                                                   const float* __restrict__ w,
                                                   const float* __restrict__ b,
                                                   unsigned short* __restrict__ o) {
  __shared__ float red[256];
  long tok = blockIdx.x;
  const float* xi = x + tok * D;
  int t = threadIdx.x;
  float s = 0.0f;
  for (int i = t; i < D; i += 256) s += xi[i];
  red[t] = s; __syncthreads();
  for (int st = 128; st > 0; st >>= 1) { if (t < st) red[t] += red[t + st]; __syncthreads(); }
  float mean = red[0] / (float)D;
  __syncthreads();
  float v = 0.0f;
  for (int i = t; i < D; i += 256) { float dd = xi[i] - mean; v += dd * dd; }
  red[t] = v; __syncthreads();
  for (int st = 128; st > 0; st >>= 1) { if (t < st) red[t] += red[t + st]; __syncthreads(); }
  float rstd = rsqrtf(red[0] / (float)D + 1e-5f);
  unsigned short* oo = o + tok * D;
  for (int i = t; i < D; i += 256) oo[i] = f2bf((xi[i] - mean) * rstd * w[i] + b[i]);
}

// causal conv over L=8 (dir=0 forward, dir=1 flipped) + SiLU; f32 + bf16 out
__global__ __launch_bounds__(256) void k_conv_silu(const float* __restrict__ xz,
                                                   const float* __restrict__ cw,
                                                   const float* __restrict__ cb,
                                                   float* __restrict__ xc,
                                                   unsigned short* __restrict__ xcb,
                                                   int dir) {
  long idx = (long)blockIdx.x * 256 + threadIdx.x;
  if (idx >= NT * D) return;
  int d = idx % D;
  long tok = idx / D;
  int l = (int)(tok & 7); long bb = tok >> 3;
  float acc = cb[d];
#pragma unroll
  for (int k = 0; k < 4; ++k) {
    int ls = dir ? (l + 3 - k) : (l - 3 + k);
    if (ls >= 0 && ls < 8) acc += xz[(bb * 8 + ls) * 1536 + d] * cw[d * 4 + k];
  }
  float r = siluf(acc);
  xc[idx] = r;
  xcb[idx] = f2bf(r);
}

// selective scan: one thread per (batch, channel); L=8, N=16
__global__ __launch_bounds__(256) void k_scan(const float* __restrict__ xc,
                                              const float* __restrict__ delta,
                                              const float* __restrict__ xdbl,
                                              const float* __restrict__ Alog,
                                              const float* __restrict__ Dp,
                                              float* __restrict__ y, int dir) {
  long idx = (long)blockIdx.x * 256 + threadIdx.x;
  if (idx >= 392L * D) return;
  int d = idx % D; long bb = idx / D;
  float Arow[16], h[16];
#pragma unroll
  for (int n = 0; n < 16; ++n) { Arow[n] = -__expf(Alog[d * 16 + n]); h[n] = 0.0f; }
  for (int s = 0; s < 8; ++s) {
    int l = dir ? (7 - s) : s;
    long tok = bb * 8 + l;
    float dlt = delta[tok * D + d];
    float u   = xc[tok * D + d];
    const float* xd = xdbl + tok * 80;
    float du = dlt * u;
    float acc = 0.0f;
#pragma unroll
    for (int n = 0; n < 16; ++n) {
      h[n] = __expf(dlt * Arow[n]) * h[n] + du * xd[48 + n];
      acc += h[n] * xd[64 + n];
    }
    y[tok * D + d] = acc + u * Dp[d];
  }
}

// (y_f + y_b) * silu(z) -> bf16 (feeds out_proj GEMM only)
__global__ __launch_bounds__(256) void k_combine(const float* __restrict__ yf,
                                                 const float* __restrict__ yb,
                                                 const float* __restrict__ xz,
                                                 unsigned short* __restrict__ yc) {
  long idx = (long)blockIdx.x * 256 + threadIdx.x;
  if (idx >= NT * D) return;
  int d = idx % D; long tok = idx / D;
  float z = xz[tok * 1536 + D + d];
  yc[idx] = f2bf((yf[idx] + yb[idx]) * siluf(z));
}

__global__ __launch_bounds__(256) void k_add(const float* __restrict__ a,
                                             const float* __restrict__ b,
                                             float* __restrict__ c, long n) {
  long idx = (long)blockIdx.x * 256 + threadIdx.x;
  if (idx < n) c[idx] = a[idx] + b[idx];
}

// xs[p=b*8+f][0]=cls(b); xs[p][1+t]=time_res[b][t][f]
__global__ __launch_bounds__(256) void k_build_xs(const float* __restrict__ x,
                                                  const float* __restrict__ tres,
                                                  float* __restrict__ xs) {
  long idx = (long)blockIdx.x * 256 + threadIdx.x;
  if (idx >= NS * D) return;
  int d = idx % D;
  long rem = idx / D;
  int t = (int)(rem % LSEQ);
  int p = (int)(rem / LSEQ);
  int b = p >> 3, f = p & 7;
  float v;
  if (t == 0) v = x[(long)b * 1569 * D + d];
  else        v = tres[(((long)b * 196 + (t - 1)) * 8 + f) * D + d];
  xs[idx] = v;
}

// xnew = res_x + concat(mean_f cls, scatter(space tokens))
__global__ __launch_bounds__(256) void k_assemble(const float* __restrict__ x,
                                                  const float* __restrict__ so,
                                                  float* __restrict__ xn) {
  long idx = (long)blockIdx.x * 256 + threadIdx.x;
  if (idx >= NX * D) return;
  int d = idx % D;
  long rem = idx / D;
  int s = (int)(rem % 1569);
  int b = (int)(rem / 1569);
  float v;
  if (s == 0) {
    float acc = 0.0f;
#pragma unroll
    for (int f = 0; f < 8; ++f) acc += so[(((long)b * 8 + f) * LSEQ + 0) * D + d];
    v = acc * 0.125f;
  } else {
    int t = (s - 1) >> 3, f = (s - 1) & 7;
    v = so[(((long)b * 8 + f) * LSEQ + 1 + t) * D + d];
  }
  xn[idx] = x[idx] + v;
}

// softmax over 197-wide rows: f32 in, bf16 out (feeds P·V GEMM)
__global__ __launch_bounds__(256) void k_softmax197(const float* __restrict__ sc,
                                                    unsigned short* __restrict__ pb) {
  __shared__ float red[256];
  long row = blockIdx.x;
  const float* p = sc + row * LSEQ;
  unsigned short* po = pb + row * LSEQ;
  int t = threadIdx.x;
  float v = (t < LSEQ) ? p[t] : -3.4e38f;
  red[t] = v; __syncthreads();
  for (int st = 128; st > 0; st >>= 1) { if (t < st) red[t] = fmaxf(red[t], red[t + st]); __syncthreads(); }
  float mx = red[0]; __syncthreads();
  float e = (t < LSEQ) ? __expf(v - mx) : 0.0f;
  red[t] = e; __syncthreads();
  for (int st = 128; st > 0; st >>= 1) { if (t < st) red[t] += red[t + st]; __syncthreads(); }
  float inv = 1.0f / red[0];
  if (t < LSEQ) po[t] = f2bf(e * inv);
}

// ---------------------------------------------------------------------------
// Host orchestration
// ---------------------------------------------------------------------------
static inline unsigned cdivu(long a, long b) { return (unsigned)((a + b - 1) / b); }

extern "C" void kernel_launch(void* const* d_in, const int* in_sizes, int n_in,
                              void* d_out, int out_size, void* d_ws, size_t ws_size,
                              hipStream_t stream) {
  (void)in_sizes; (void)n_in; (void)out_size; (void)ws_size;
  const float* x        = (const float*)d_in[0];
  const float* n1w      = (const float*)d_in[1];
  const float* n1b      = (const float*)d_in[2];
  const float* n2w      = (const float*)d_in[3];
  const float* n2b      = (const float*)d_in[4];
  const float* n3w      = (const float*)d_in[5];
  const float* n3b      = (const float*)d_in[6];
  const float* attn_in_w  = (const float*)d_in[7];
  const float* attn_in_b  = (const float*)d_in[8];
  const float* attn_out_w = (const float*)d_in[9];
  const float* attn_out_b = (const float*)d_in[10];
  const float* m_in_w   = (const float*)d_in[11];
  const float* m_cw     = (const float*)d_in[12];
  const float* m_cb     = (const float*)d_in[13];
  const float* m_xw     = (const float*)d_in[14];
  const float* m_dw     = (const float*)d_in[15];
  const float* m_db     = (const float*)d_in[16];
  const float* m_Al     = (const float*)d_in[17];
  const float* m_Dp     = (const float*)d_in[18];
  const float* m_cwb    = (const float*)d_in[19];
  const float* m_cbb    = (const float*)d_in[20];
  const float* m_xwb    = (const float*)d_in[21];
  const float* m_dwb    = (const float*)d_in[22];
  const float* m_dbb    = (const float*)d_in[23];
  const float* m_Alb    = (const float*)d_in[24];
  const float* m_Dpb    = (const float*)d_in[25];
  const float* m_ow     = (const float*)d_in[26];
  const float* fc1_w    = (const float*)d_in[27];
  const float* fc1_b    = (const float*)d_in[28];
  const float* fc2_w    = (const float*)d_in[29];
  const float* fc2_b    = (const float*)d_in[30];

  char* wsb  = (char*)d_ws;
  float* out = (float*)d_out;
  auto F  = [&](size_t o) { return (float*)(wsb + o); };
  auto U  = [&](size_t o) { return (unsigned short*)(wsb + o); };
  dim3 blk(256);

  // ---- pre-convert weights to bf16 (once per call) ----
  struct { const float* src; size_t dst; long n; } wc[] = {
    { m_in_w,     B_WIN,  1536L * 768 }, { m_xw,  B_WXF, 80L * 768 },
    { m_xwb,      B_WXB,  80L * 768 },   { m_dw,  B_WDF, 768L * 48 },
    { m_dwb,      B_WDB,  768L * 48 },   { m_ow,  B_WOW, 768L * 768 },
    { attn_in_w,  B_WQKV, 2304L * 768 }, { attn_out_w, B_WAO, 768L * 768 },
    { fc1_w,      B_WFC1, 3072L * 768 }, { fc2_w, B_WFC2, 768L * 3072 },
  };
  for (auto& w : wc)
    k_cvt<<<cdivu(w.n, 256), blk, 0, stream>>>(w.src, U(w.dst), w.n);

  // ---- time-Mamba branch ----
  k_gather_xt<<<cdivu(NT * D, 256), blk, 0, stream>>>(x, F(B_XT));
  k_layernorm<<<(unsigned)NT, blk, 0, stream>>>(F(B_XT), n3w, n3b, U(B_LN3B));

  // in_proj: [NT,768] x [1536,768]^T -> xz [NT,1536] f32
  k_gemm<0, true, 0><<<dim3(cdivu(1536, TMM), cdivu(NT, TMM), 1), blk, 0, stream>>>(
      U(B_LN3B), 768, 0, 0, U(B_WIN), 768, 0, 0, F(B_XZ), nullptr, 1536, 0, 0,
      nullptr, (int)NT, 1536, 768, 1, 1.0f);

  k_conv_silu<<<cdivu(NT * D, 256), blk, 0, stream>>>(F(B_XZ), m_cw,  m_cb,  F(B_XCF), U(B_XCFB), 0);
  k_conv_silu<<<cdivu(NT * D, 256), blk, 0, stream>>>(F(B_XZ), m_cwb, m_cbb, F(B_XCB), U(B_XCBB), 1);

  // x_proj: [NT,768] x [80,768]^T -> [NT,80] f32 + bf16
  k_gemm<0, true, 2><<<dim3(1, cdivu(NT, TMM), 1), blk, 0, stream>>>(
      U(B_XCFB), 768, 0, 0, U(B_WXF), 768, 0, 0, F(B_XDF), U(B_XDFB), 80, 0, 0,
      nullptr, (int)NT, 80, 768, 1, 1.0f);
  k_gemm<0, true, 2><<<dim3(1, cdivu(NT, TMM), 1), blk, 0, stream>>>(
      U(B_XCBB), 768, 0, 0, U(B_WXB), 768, 0, 0, F(B_XDB), U(B_XDBB), 80, 0, 0,
      nullptr, (int)NT, 80, 768, 1, 1.0f);

  // delta = softplus(dt @ dt_w^T + dt_b): [NT,48] x [768,48]^T -> [NT,768] f32
  k_gemm<3, true, 0><<<dim3(cdivu(768, TMM), cdivu(NT, TMM), 1), blk, 0, stream>>>(
      U(B_XDFB), 80, 0, 0, U(B_WDF), 48, 0, 0, F(B_DLF), nullptr, 768, 0, 0,
      m_db, (int)NT, 768, 48, 1, 1.0f);
  k_gemm<3, true, 0><<<dim3(cdivu(768, TMM), cdivu(NT, TMM), 1), blk, 0, stream>>>(
      U(B_XDBB), 80, 0, 0, U(B_WDB), 48, 0, 0, F(B_DLB), nullptr, 768, 0, 0,
      m_dbb, (int)NT, 768, 48, 1, 1.0f);

  k_scan<<<cdivu(392L * D, 256), blk, 0, stream>>>(F(B_XCF), F(B_DLF), F(B_XDF), m_Al,  m_Dp,  F(B_YF), 0);
  k_scan<<<cdivu(392L * D, 256), blk, 0, stream>>>(F(B_XCB), F(B_DLB), F(B_XDB), m_Alb, m_Dpb, F(B_YB), 1);
  k_combine<<<cdivu(NT * D, 256), blk, 0, stream>>>(F(B_YF), F(B_YB), F(B_XZ), U(B_YCB));

  // out_proj: [NT,768] x [768,768]^T -> f32
  k_gemm<0, true, 0><<<dim3(cdivu(768, TMM), cdivu(NT, TMM), 1), blk, 0, stream>>>(
      U(B_YCB), 768, 0, 0, U(B_WOW), 768, 0, 0, F(B_TOUT), nullptr, 768, 0, 0,
      nullptr, (int)NT, 768, 768, 1, 1.0f);
  k_add<<<cdivu(NT * D, 256), blk, 0, stream>>>(F(B_XT), F(B_TOUT), F(B_TRES), NT * D);

  // ---- space-attention branch ----
  k_build_xs<<<cdivu(NS * D, 256), blk, 0, stream>>>(x, F(B_TRES), F(B_XS));
  k_layernorm<<<(unsigned)NS, blk, 0, stream>>>(F(B_XS), n1w, n1b, U(B_LN1B));

  // QKV: [NS,768] x [2304,768]^T + bias -> bf16 [NS,2304]
  k_gemm<1, true, 1><<<dim3(cdivu(2304, TMM), cdivu(NS, TMM), 1), blk, 0, stream>>>(
      U(B_LN1B), 768, 0, 0, U(B_WQKV), 768, 0, 0, nullptr, U(B_QKVB), 2304, 0, 0,
      attn_in_b, (int)NS, 2304, 768, 1, 1.0f);

  // scores = (Q K^T) / 8 per (pair,head) -> f32
  k_gemm<0, true, 0><<<dim3(cdivu(LSEQ, TMM), cdivu(LSEQ, TMM), NPAIR * NHEAD), blk, 0, stream>>>(
      U(B_QKVB),        2304, LSEQ * 2304, HD,
      U(B_QKVB) + 768,  2304, LSEQ * 2304, HD,
      F(B_SC), nullptr, (int)LSEQ, NHEAD * LSEQ * LSEQ, LSEQ * LSEQ,
      nullptr, (int)LSEQ, (int)LSEQ, (int)HD, (int)NHEAD, 0.125f);

  k_softmax197<<<(unsigned)(192L * LSEQ), blk, 0, stream>>>(F(B_SC), U(B_PB));

  // O = P V per (pair,head): B stored [K=197][N=64] -> bf16
  k_gemm<0, false, 1><<<dim3(1, cdivu(LSEQ, TMM), NPAIR * NHEAD), blk, 0, stream>>>(
      U(B_PB), (int)LSEQ, NHEAD * LSEQ * LSEQ, LSEQ * LSEQ,
      U(B_QKVB) + 1536, 2304, LSEQ * 2304, HD,
      nullptr, U(B_AOB), 768, LSEQ * 768, HD,
      nullptr, (int)LSEQ, (int)HD, (int)LSEQ, (int)NHEAD, 1.0f);

  // attn out-proj -> f32
  k_gemm<1, true, 0><<<dim3(cdivu(768, TMM), cdivu(NS, TMM), 1), blk, 0, stream>>>(
      U(B_AOB), 768, 0, 0, U(B_WAO), 768, 0, 0, F(B_SOUT), nullptr, 768, 0, 0,
      attn_out_b, (int)NS, 768, 768, 1, 1.0f);

  k_assemble<<<cdivu(NX * D, 256), blk, 0, stream>>>(x, F(B_SOUT), F(B_XNEW));

  // ---- MLP ----
  k_layernorm<<<(unsigned)NX, blk, 0, stream>>>(F(B_XNEW), n2w, n2b, U(B_LN2B));
  k_gemm<2, true, 1><<<dim3(cdivu(3072, TMM), cdivu(NX, TMM), 1), blk, 0, stream>>>(
      U(B_LN2B), 768, 0, 0, U(B_WFC1), 768, 0, 0, nullptr, U(B_H1B), 3072, 0, 0,
      fc1_b, (int)NX, 3072, 768, 1, 1.0f);
  k_gemm<1, true, 0><<<dim3(cdivu(768, TMM), cdivu(NX, TMM), 1), blk, 0, stream>>>(
      U(B_H1B), 3072, 0, 0, U(B_WFC2), 3072, 0, 0, F(B_H2), nullptr, 768, 0, 0,
      fc2_b, (int)NX, 768, 3072, 1, 1.0f);

  k_add<<<cdivu(NX * D, 256), blk, 0, stream>>>(F(B_XNEW), F(B_H2), out, NX * D);
}
